// MambaMixer_25494925869696
// MI455X (gfx1250) — matrix operand order repaired and checked by reading, not compiled, not run on hardware
//
#include <hip/hip_runtime.h>
#include <hip/hip_bf16.h>

// ---------------- CDNA5 WMMA types / helpers ----------------
typedef __attribute__((ext_vector_type(16))) __bf16 v16bf;
typedef __attribute__((ext_vector_type(8)))  float  v8f;
typedef __attribute__((ext_vector_type(4)))  __bf16 v4bf;
typedef __attribute__((ext_vector_type(4)))  float  f32x4;

__device__ __forceinline__ v8f wmma_bf16(v16bf a, v16bf b, v8f c) {
    // D = A(16x32) * B(32x16) + C(16x16), f32 accumulate
    return __builtin_amdgcn_wmma_f32_16x16x32_bf16(
        false, a, false, b, (short)0, c, false, false);
}

// Padded LDS row stride (bf16 elements) for 32-wide K tiles:
// 40*2 = 80B per row -> 20-bank stride, fragment chunks stay 16B aligned.
#define SK 40

// A-fragment: element j at k = (j/8)*16 + khalf*8 + (j%8)  (two 16B chunks)
__device__ __forceinline__ v16bf frag_a_row(const __bf16* row, int kh) {
    v16bf r;
#pragma unroll
    for (int j = 0; j < 16; ++j) {
        int k = ((j >> 3) << 4) + (kh << 3) + (j & 7);
        r[j] = row[k];
    }
    return r;
}
// B-fragment: element j at k = (j/8)*16 + khalf*8 + (j%8)  (two 16B chunks)
__device__ __forceinline__ v16bf frag_b_row(const __bf16* row, int kh) {
    v16bf r;
#pragma unroll
    for (int j = 0; j < 16; ++j) r[j] = row[((j >> 3) << 4) + (kh << 3) + (j & 7)];
    return r;
}

// ---------------- Problem constants ----------------
#define LSEQ 2048
#define HIDD 2048
#define II   4096
#define HH   64
#define PP   64
#define NN   128
#define QQ   256
#define NC   8
#define CDIM 4352         // I + 2*G*N
#define OPROJ 8512        // I + CDIM + H

// ---------------- Generic batched WMMA GEMM ----------------
// C[m][n] = sum_k A[m][k] * B[n][k]   (both row-major f32; "A x B^T")
#define BM 128
#define BN 128
#define BK 32

// Stage a 128x32 f32 tile -> bf16 LDS tile (rows padded to SK).
// 256 threads, 4 x float4 each; loads issued before converts.
__device__ __forceinline__ void stage_full(const float* __restrict__ src, int ld,
                                           __bf16* __restrict__ dst, int tid) {
    f32x4 tmp[4];
#pragma unroll
    for (int t = 0; t < 4; ++t) {
        int fi = t * 256 + tid;
        int r = fi >> 3, c4 = (fi & 7) << 2;
        tmp[t] = *(const f32x4*)(src + (long long)r * ld + c4);
    }
#pragma unroll
    for (int t = 0; t < 4; ++t) {
        int fi = t * 256 + tid;
        int r = fi >> 3, c4 = (fi & 7) << 2;
        v4bf o;
        o[0] = (__bf16)tmp[t][0]; o[1] = (__bf16)tmp[t][1];
        o[2] = (__bf16)tmp[t][2]; o[3] = (__bf16)tmp[t][3];
        *(v4bf*)(dst + r * SK + c4) = o;
    }
}
__device__ __forceinline__ void stage_guard(const float* __restrict__ src, int ld, int rows,
                                            __bf16* __restrict__ dst, int tid) {
    f32x4 tmp[4];
#pragma unroll
    for (int t = 0; t < 4; ++t) {
        int fi = t * 256 + tid;
        int r = fi >> 3, c4 = (fi & 7) << 2;
        f32x4 v = {0.f, 0.f, 0.f, 0.f};
        if (r < rows) v = *(const f32x4*)(src + (long long)r * ld + c4);
        tmp[t] = v;
    }
#pragma unroll
    for (int t = 0; t < 4; ++t) {
        int fi = t * 256 + tid;
        int r = fi >> 3, c4 = (fi & 7) << 2;
        v4bf o;
        o[0] = (__bf16)tmp[t][0]; o[1] = (__bf16)tmp[t][1];
        o[2] = (__bf16)tmp[t][2]; o[3] = (__bf16)tmp[t][3];
        *(v4bf*)(dst + r * SK + c4) = o;
    }
}

__global__ __launch_bounds__(256) void gemm_bf16_wmma(
    const float* __restrict__ A, const float* __restrict__ B, float* __restrict__ C,
    int M, int N, int K, int lda, int ldb, int ldc,
    long long sA, long long sB, long long sC)
{
    __shared__ __bf16 As[2][BM * SK];
    __shared__ __bf16 Bs[2][BN * SK];

    const int bz = blockIdx.z;
    A += (long long)bz * sA; B += (long long)bz * sB; C += (long long)bz * sC;
    const int bn = blockIdx.x * BN;
    const int bm = blockIdx.y * BM;
    const int tid  = threadIdx.x;
    const int lane = tid & 31;
    const int wave = tid >> 5;
    const int wm = wave >> 2;          // 0..1  -> 64 rows of M
    const int wn = wave & 3;           // 0..3  -> 32 cols of N
    const int ln = lane & 15;
    const int kh = lane >> 4;

    // M and K are always tile multiples in our launches; only N can have a tail.
    const bool fullN = (bn + BN <= N);
    const float* Ab = A + (long long)bm * lda;
    const float* Bb = B + (long long)bn * ldb;

    v8f acc[4][2];
#pragma unroll
    for (int i = 0; i < 4; ++i)
#pragma unroll
        for (int j = 0; j < 2; ++j) acc[i][j] = (v8f){0,0,0,0,0,0,0,0};

    // prologue: stage k0 = 0
    stage_full(Ab, lda, As[0], tid);
    if (fullN) stage_full(Bb, ldb, Bs[0], tid);
    else       stage_guard(Bb, ldb, N - bn, Bs[0], tid);
    __syncthreads();

    int buf = 0;
    for (int k0 = 0; k0 < K; k0 += BK) {
        // stage next K tile into the other buffer (overlaps with WMMA below)
        if (k0 + BK < K) {
            stage_full(Ab + k0 + BK, lda, As[buf ^ 1], tid);
            if (fullN) stage_full(Bb + k0 + BK, ldb, Bs[buf ^ 1], tid);
            else       stage_guard(Bb + k0 + BK, ldb, N - bn, Bs[buf ^ 1], tid);
        }
        const __bf16* Ap_ = As[buf];
        const __bf16* Bp_ = Bs[buf];
        v16bf af[4], bfr[2];
#pragma unroll
        for (int i = 0; i < 4; ++i)
            af[i] = frag_a_row(Ap_ + (wm * 64 + i * 16 + ln) * SK, kh);
#pragma unroll
        for (int j = 0; j < 2; ++j)
            bfr[j] = frag_b_row(Bp_ + (wn * 32 + j * 16 + ln) * SK, kh);
#pragma unroll
        for (int i = 0; i < 4; ++i)
#pragma unroll
            for (int j = 0; j < 2; ++j)
                acc[i][j] = wmma_bf16(af[i], bfr[j], acc[i][j]);
        __syncthreads();
        buf ^= 1;
    }

#pragma unroll
    for (int i = 0; i < 4; ++i)
#pragma unroll
        for (int j = 0; j < 2; ++j) {
            int n = bn + wn * 32 + j * 16 + ln;
            if (n < N) {
#pragma unroll
                for (int r = 0; r < 8; ++r) {
                    int m = bm + wm * 64 + i * 16 + kh * 8 + r;
                    C[(long long)m * ldc + n] = acc[i][j][r];
                }
            }
        }
}

// ---------------- conv (depthwise K=4) + SiLU ----------------
__global__ void conv_silu_kernel(const float* __restrict__ proj,
                                 const float* __restrict__ convw,
                                 const float* __restrict__ convb,
                                 float* __restrict__ xBC)
{
    long long idx = (long long)blockIdx.x * 256 + threadIdx.x;
    if (idx >= (long long)LSEQ * CDIM) return;
    int l  = (int)(idx / CDIM);
    int cc = (int)(idx % CDIM);
    float acc = convb[cc];
#pragma unroll
    for (int k = 0; k < 4; ++k) {
        int lk = l + k - 3;
        if (lk >= 0) acc += proj[(long long)lk * OPROJ + II + cc] * convw[cc * 4 + k];
    }
    xBC[idx] = acc / (1.f + __expf(-acc));   // silu
}

// ---------------- dt = softplus(proj_dt + bias) ----------------
__global__ void dt_softplus_kernel(const float* __restrict__ proj,
                                   const float* __restrict__ dt_bias,
                                   float* __restrict__ dtb)
{
    int idx = blockIdx.x * 256 + threadIdx.x;
    if (idx >= LSEQ * HH) return;
    int l = idx >> 6, h = idx & 63;
    float v = proj[(long long)l * OPROJ + (II + CDIM) + h] + dt_bias[h];
    dtb[idx] = (v > 20.f) ? v : log1pf(__expf(v));
}

// ---------------- dA cumsum per (chunk, head) ----------------
__global__ __launch_bounds__(256) void dacs_kernel(const float* __restrict__ dtb,
                                                   const float* __restrict__ A,
                                                   float* __restrict__ dAcs)
{
    int h = blockIdx.x, c = blockIdx.y;
    int tid = threadIdx.x;
    __shared__ float sv[256];
    sv[tid] = dtb[(c * QQ + tid) * HH + h] * A[h];
    __syncthreads();
    for (int off = 1; off < 256; off <<= 1) {
        float t = (tid >= off) ? sv[tid - off] : 0.f;
        __syncthreads();
        sv[tid] += t;
        __syncthreads();
    }
    dAcs[((long long)(c * HH + h)) * QQ + tid] = sv[tid];
}

// ---------------- intra-chunk y + chunk states (WMMA) ----------------
// block = (head h, chunk c), 8 waves
// y_intra[q,p] = sum_s (s<=q) CB[q,s]*exp(dA[q]-dA[s])*dt[s] * x[s,p]   (+ D*x)
// states[p,n]  = sum_q exp(dA_last - dA[q])*dt[q] * x[q,p] * B[q,n]
__global__ __launch_bounds__(256) void intra_states_kernel(
    const float* __restrict__ xBC, const float* __restrict__ dtb,
    const float* __restrict__ dAcs, const float* __restrict__ CB,
    const float* __restrict__ Dp, float* __restrict__ y, float* __restrict__ states)
{
    const int h = blockIdx.x;
    const int c = blockIdx.y;
    const int tid  = threadIdx.x;
    const int lane = tid & 31;
    const int wave = tid >> 5;
    const int ln = lane & 15;
    const int kh = lane >> 4;

    __shared__ float s_dt[QQ], s_dA[QQ], s_w[QQ];
    __shared__ __bf16 s_xT[PP * SK];     // x^T tile   [p][s]
    __shared__ __bf16 s_wxT[PP * SK];    // (w*x)^T    [p][s]
    __shared__ __bf16 s_BT[NN * SK];     // B^T tile   [n][s]

    const long long chb = ((long long)c * HH + h) * QQ;
    s_dt[tid] = dtb[(long long)(c * QQ + tid) * HH + h];
    s_dA[tid] = dAcs[chb + tid];
    __syncthreads();
    const float dAlast = s_dA[QQ - 1];
    s_w[tid] = __expf(dAlast - s_dA[tid]) * s_dt[tid];
    __syncthreads();

    v8f accY[2][4], accS[4];
#pragma unroll
    for (int a = 0; a < 2; ++a)
#pragma unroll
        for (int b = 0; b < 4; ++b) accY[a][b] = (v8f){0,0,0,0,0,0,0,0};
#pragma unroll
    for (int b = 0; b < 4; ++b) accS[b] = (v8f){0,0,0,0,0,0,0,0};

    const long long rowb = (long long)c * QQ * CDIM;
    const long long cbb  = (long long)c * QQ * QQ;

    for (int st = 0; st < 8; ++st) {
        const int s0 = st * 32;
        // stage x^T and (w*x)^T (64p x 32s): lanes walk p (coalesced),
        // each thread packs 4 s-consecutive bf16 -> one ds_store_b64
#pragma unroll
        for (int t = 0; t < 2; ++t) {
            int fi = t * 256 + tid;
            int p = fi & 63, s4 = (fi >> 6) << 2;
            float xv[4];
#pragma unroll
            for (int u = 0; u < 4; ++u)
                xv[u] = xBC[rowb + (long long)(s0 + s4 + u) * CDIM + h * PP + p];
            v4bf xo, wo;
#pragma unroll
            for (int u = 0; u < 4; ++u) {
                xo[u] = (__bf16)xv[u];
                wo[u] = (__bf16)(xv[u] * s_w[s0 + s4 + u]);
            }
            *(v4bf*)(s_xT  + p * SK + s4) = xo;
            *(v4bf*)(s_wxT + p * SK + s4) = wo;
        }
        // stage B^T (128n x 32s)
#pragma unroll
        for (int t = 0; t < 4; ++t) {
            int fi = t * 256 + tid;
            int n = fi & 127, s4 = (fi >> 7) << 2;
            float bv[4];
#pragma unroll
            for (int u = 0; u < 4; ++u)
                bv[u] = xBC[rowb + (long long)(s0 + s4 + u) * CDIM + II + n];
            v4bf o;
#pragma unroll
            for (int u = 0; u < 4; ++u) o[u] = (__bf16)bv[u];
            *(v4bf*)(s_BT + n * SK + s4) = o;
        }
        __syncthreads();

        // ---- states: wave w covers p-tile (w>>1), n-half (w&1)
        {
            int pt = wave >> 1;
            v16bf af = frag_a_row(s_wxT + (pt * 16 + ln) * SK, kh);
#pragma unroll
            for (int j = 0; j < 4; ++j) {
                int n = ((wave & 1) * 4 + j) * 16 + ln;
                v16bf bfg = frag_b_row(s_BT + n * SK, kh);
                accS[j] = wmma_bf16(af, bfg, accS[j]);
            }
        }
        // ---- y_intra: wave w covers q in [w*32, w*32+32); skip above-diagonal tiles
        if (s0 <= wave * 32 + 31) {
            v16bf xfrag[4];
#pragma unroll
            for (int pt = 0; pt < 4; ++pt)
                xfrag[pt] = frag_b_row(s_xT + (pt * 16 + ln) * SK, kh);
#pragma unroll
            for (int qt = 0; qt < 2; ++qt) {
                int q = wave * 32 + qt * 16 + ln;
                float dq = s_dA[q];
                v16bf mf;
#pragma unroll
                for (int j = 0; j < 16; ++j) {
                    int sl = ((j >> 3) << 4) + (kh << 3) + (j & 7);
                    int s = s0 + sl;
                    float mv = 0.f;
                    if (s <= q)
                        mv = CB[cbb + q * QQ + s] * __expf(dq - s_dA[s]) * s_dt[s];
                    mf[j] = (__bf16)mv;
                }
#pragma unroll
                for (int pt = 0; pt < 4; ++pt)
                    accY[qt][pt] = wmma_bf16(mf, xfrag[pt], accY[qt][pt]);
            }
        }
        __syncthreads();
    }

    // write y_intra + D*x
    const float Dh = Dp[h];
#pragma unroll
    for (int qt = 0; qt < 2; ++qt)
#pragma unroll
        for (int pt = 0; pt < 4; ++pt) {
            int p = pt * 16 + ln;
#pragma unroll
            for (int r = 0; r < 8; ++r) {
                int q = wave * 32 + qt * 16 + kh * 8 + r;
                float xv = xBC[rowb + (long long)q * CDIM + h * PP + p];
                y[((long long)(c * QQ + q)) * II + h * PP + p] = accY[qt][pt][r] + Dh * xv;
            }
        }
    // write states[c][h][p][n]
    const long long stb = ((long long)(c * HH + h)) * PP * NN;
    {
        int pt = wave >> 1;
#pragma unroll
        for (int j = 0; j < 4; ++j) {
            int n = ((wave & 1) * 4 + j) * 16 + ln;
#pragma unroll
            for (int r = 0; r < 8; ++r) {
                int p = pt * 16 + kh * 8 + r;
                states[stb + (long long)p * NN + n] = accS[j][r];
            }
        }
    }
}

// ---------------- inter-chunk scan + y_inter (WMMA) ----------------
// One block per head; S (64x128 f32, rows padded to 132) resident in LDS.
#define SNPAD 132
__global__ __launch_bounds__(256) void inter_kernel(
    const float* __restrict__ xBC, const float* __restrict__ dAcs,
    const float* __restrict__ states, float* __restrict__ y)
{
    const int h = blockIdx.x;
    const int tid  = threadIdx.x;
    const int lane = tid & 31;
    const int wave = tid >> 5;
    const int ln = lane & 15;
    const int kh = lane >> 4;

    __shared__ float  S[PP * SNPAD];    // running state, f32
    __shared__ __bf16 Cs[QQ * SK];      // C chunk k-tile
    __shared__ float  s_scale[QQ];

    for (int i = tid; i < PP * SNPAD; i += 256) S[i] = 0.f;

    for (int c = 0; c < NC; ++c) {
        const long long chb = ((long long)c * HH + h) * QQ;
        s_scale[tid] = __expf(dAcs[chb + tid]);
        __syncthreads();

        v8f acc[2][4];
#pragma unroll
        for (int a = 0; a < 2; ++a)
#pragma unroll
            for (int b = 0; b < 4; ++b) acc[a][b] = (v8f){0,0,0,0,0,0,0,0};

        for (int n0 = 0; n0 < NN; n0 += 32) {
            // stage C tile (128q x 32n), vectorized f32x4 loads + b64 LDS stores
#pragma unroll
            for (int t = 0; t < 4; ++t) {
                int fi = t * 256 + tid;
                int q = fi >> 3, n4 = (fi & 7) << 2;
                f32x4 v = *(const f32x4*)(xBC + ((long long)(c * QQ + q)) * CDIM
                                          + II + NN + n0 + n4);
                v4bf o;
                o[0] = (__bf16)v[0]; o[1] = (__bf16)v[1];
                o[2] = (__bf16)v[2]; o[3] = (__bf16)v[3];
                *(v4bf*)(Cs + q * SK + n4) = o;
            }
            __syncthreads();
            // B-fragments from S (built once per K tile, reused by both q rows)
            v16bf sfr[4];
#pragma unroll
            for (int pt = 0; pt < 4; ++pt) {
                const float* srow = S + (pt * 16 + ln) * SNPAD + n0;
                v16bf bfg;
#pragma unroll
                for (int j = 0; j < 16; ++j) bfg[j] = (__bf16)srow[((j >> 3) << 4) + (kh << 3) + (j & 7)];
                sfr[pt] = bfg;
            }
#pragma unroll
            for (int qt = 0; qt < 2; ++qt) {
                v16bf af = frag_a_row(Cs + (wave * 32 + qt * 16 + ln) * SK, kh);
#pragma unroll
                for (int pt = 0; pt < 4; ++pt)
                    acc[qt][pt] = wmma_bf16(af, sfr[pt], acc[qt][pt]);
            }
            __syncthreads();
        }

        // y += exp(dA_cs[q]) * (C @ S^T)
#pragma unroll
        for (int qt = 0; qt < 2; ++qt)
#pragma unroll
            for (int pt = 0; pt < 4; ++pt) {
                int p = pt * 16 + ln;
#pragma unroll
                for (int r = 0; r < 8; ++r) {
                    int q = wave * 32 + qt * 16 + kh * 8 + r;
                    long long yi = ((long long)(c * QQ + q)) * II + h * PP + p;
                    y[yi] += s_scale[q] * acc[qt][pt][r];
                }
            }

        // S <- chunk_decay * S + states[c]
        const float cd = __expf(dAcs[chb + QQ - 1]);
        const long long stb = ((long long)(c * HH + h)) * PP * NN;
        for (int i = tid; i < PP * NN; i += 256) {
            int p = i >> 7, n = i & 127;
            S[p * SNPAD + n] = cd * S[p * SNPAD + n] + states[stb + i];
        }
        __syncthreads();
    }
}

// ---------------- gate (SiLU) + RMS-style norm ----------------
__global__ __launch_bounds__(256) void gate_norm_kernel(
    const float* __restrict__ proj, const float* __restrict__ normw,
    float* __restrict__ y)
{
    const int l = blockIdx.x;
    const int tid = threadIdx.x;
    __shared__ float red[256];
    float g[16];
    float ss = 0.f;
#pragma unroll
    for (int t = 0; t < 16; ++t) {
        int i = t * 256 + tid;
        float gate = proj[(long long)l * OPROJ + i];
        float yv   = y[(long long)l * II + i];
        float gv   = yv * (gate / (1.f + __expf(-gate)));
        g[t] = gv;
        ss += gv * gv;
    }
    red[tid] = ss;
    __syncthreads();
    for (int off = 128; off > 0; off >>= 1) {
        if (tid < off) red[tid] += red[tid + off];
        __syncthreads();
    }
    float inv = rsqrtf(red[0] / (float)II + 1e-5f);
#pragma unroll
    for (int t = 0; t < 16; ++t) {
        int i = t * 256 + tid;
        y[(long long)l * II + i] = g[t] * inv * normw[i];
    }
}

// ---------------- workspace layout (floats) ----------------
static constexpr long long SZ_PROJ = (long long)LSEQ * OPROJ;     // 17,432,576
static constexpr long long SZ_XBC  = (long long)LSEQ * CDIM;      //  8,912,896
static constexpr long long SZ_DT   = (long long)LSEQ * HH;        //    131,072
static constexpr long long SZ_DACS = (long long)NC * HH * QQ;     //    131,072
static constexpr long long SZ_CB   = (long long)NC * QQ * QQ;     //    524,288
static constexpr long long SZ_Y    = (long long)LSEQ * II;        //  8,388,608
static constexpr long long OFF_PROJ = 0;
static constexpr long long OFF_XBC  = OFF_PROJ + SZ_PROJ;
static constexpr long long OFF_DT   = OFF_XBC + SZ_XBC;
static constexpr long long OFF_DACS = OFF_DT + SZ_DT;
static constexpr long long OFF_CB   = OFF_DACS + SZ_DACS;
static constexpr long long OFF_Y    = OFF_CB + SZ_CB;
static constexpr long long OFF_ST   = OFF_Y + SZ_Y;

extern "C" void kernel_launch(void* const* d_in, const int* in_sizes, int n_in,
                              void* d_out, int out_size, void* d_ws, size_t ws_size,
                              hipStream_t stream) {
    const float* hidden     = (const float*)d_in[0];
    const float* in_proj_w  = (const float*)d_in[1];
    const float* conv_w     = (const float*)d_in[2];
    const float* conv_b     = (const float*)d_in[3];
    const float* dt_bias    = (const float*)d_in[4];
    const float* Ap         = (const float*)d_in[5];
    const float* Dp         = (const float*)d_in[6];
    const float* norm_w     = (const float*)d_in[7];
    const float* out_proj_w = (const float*)d_in[8];
    float* out = (float*)d_out;
    float* w   = (float*)d_ws;

    // 1. in_proj: proj[l,o] = hidden[l,:] . W[o,:]
    gemm_bf16_wmma<<<dim3((OPROJ + BN - 1) / BN, LSEQ / BM, 1), 256, 0, stream>>>(
        hidden, in_proj_w, w + OFF_PROJ,
        LSEQ, OPROJ, HIDD, HIDD, HIDD, OPROJ, 0, 0, 0);

    // 2. conv + SiLU, dt softplus
    conv_silu_kernel<<<(int)(((long long)LSEQ * CDIM + 255) / 256), 256, 0, stream>>>(
        w + OFF_PROJ, conv_w, conv_b, w + OFF_XBC);
    dt_softplus_kernel<<<(LSEQ * HH + 255) / 256, 256, 0, stream>>>(
        w + OFF_PROJ, dt_bias, w + OFF_DT);

    // 3. dA cumsum
    dacs_kernel<<<dim3(HH, NC), 256, 0, stream>>>(w + OFF_DT, Ap, w + OFF_DACS);

    // 4. CB[q,s] = C[q,:].B[s,:]   (shared across heads, G=1) — batched over chunks
    gemm_bf16_wmma<<<dim3(QQ / BN, QQ / BM, NC), 256, 0, stream>>>(
        w + OFF_XBC + II + NN, w + OFF_XBC + II, w + OFF_CB,
        QQ, QQ, NN, CDIM, CDIM, QQ,
        (long long)QQ * CDIM, (long long)QQ * CDIM, (long long)QQ * QQ);

    // 5. intra-chunk y + chunk states
    intra_states_kernel<<<dim3(HH, NC), 256, 0, stream>>>(
        w + OFF_XBC, w + OFF_DT, w + OFF_DACS, w + OFF_CB, Dp, w + OFF_Y, w + OFF_ST);

    // 6. inter-chunk scan + y_inter
    inter_kernel<<<HH, 256, 0, stream>>>(w + OFF_XBC, w + OFF_DACS, w + OFF_ST, w + OFF_Y);

    // 7. gate + norm (in place on y)
    gate_norm_kernel<<<LSEQ, 256, 0, stream>>>(w + OFF_PROJ, norm_w, w + OFF_Y);

    // 8. out_proj -> d_out
    gemm_bf16_wmma<<<dim3(HIDD / BN, LSEQ / BM, 1), 256, 0, stream>>>(
        w + OFF_Y, out_proj_w, out,
        LSEQ, HIDD, II, II, II, HIDD, 0, 0, 0);
}
